// f_r_36910948942540
// MI455X (gfx1250) — compile-verified, matrix operand order checked by reading, not run on hardware
//
#include <hip/hip_runtime.h>
#include <hip/hip_bf16.h>

typedef __attribute__((ext_vector_type(16))) _Float16 v16h;
typedef __attribute__((ext_vector_type(8)))  _Float16 v8h;
typedef __attribute__((ext_vector_type(2)))  _Float16 v2h;
typedef __attribute__((ext_vector_type(8)))  float    v8f;

#define U3        384              // 3*U
#define D_IN      128
#define T_HIST    8
#define NTILES    24               // 384/16
#define FRAG_HALVES (NTILES*4*32*16)   // 49152 halves per swizzled matrix
#define GRU_THREADS 384            // 12 waves * 32

// ---------------------------------------------------------------------------
// Weight pre-swizzle: fp32 [128 x 384] -> f16 in exact WMMA B-fragment order.
// Fragment f = ct*4 + ks (ct = N-tile 0..23, ks = K-step 0..3).
// Lane l holds column n = ct*16 + (l&15); element e holds
//   k = ks*32 + ((l>=16)?8:0) + (e<8 ? e : e+8)      (16-bit B layout, wave32)
// ---------------------------------------------------------------------------
__global__ void swizzle_weights(const float* eK, const float* eR,
                                const float* vK, const float* vR,
                                const float* uK, const float* uR,
                                _Float16* dst) {
    int i = blockIdx.x * blockDim.x + threadIdx.x;
    if (i >= 6 * FRAG_HALVES) return;
    int mat = i / FRAG_HALVES;
    int r   = i % FRAG_HALVES;
    int e  = r & 15;
    int l  = (r >> 4) & 31;
    int f  = r >> 9;                  // 0..95
    int ks = f & 3;
    int ct = f >> 2;
    int n  = ct * 16 + (l & 15);
    int k  = ks * 32 + ((l >> 4) ? 8 : 0) + (e < 8 ? e : e + 8);
    const float* src;
    if      (mat == 0) src = eK;
    else if (mat == 1) src = eR;
    else if (mat == 2) src = vK;
    else if (mat == 3) src = vR;
    else if (mat == 4) src = uK;
    else               src = uR;
    dst[i] = (_Float16)src[(size_t)k * U3 + n];
}

__global__ void zero_f32(float* p, int n) {
    int i = blockIdx.x * blockDim.x + threadIdx.x;
    if (i < n) p[i] = 0.f;
}

__device__ __forceinline__ float sigmoidf_(float x) {
    return 1.f / (1.f + __expf(-x));
}

// A-fragment half-index for element (m, k), k in [0,128):
//   ks = k>>5; kk = k&31; lane = m + ((kk&8)?16:0); e = (kk&7) | ((kk&16)>>1)
//   index = (ks*32 + lane)*16 + e
__device__ __forceinline__ int afrag_idx(int m, int k) {
    int ks = k >> 5;
    int kk = k & 31;
    int lane = m + ((kk & 8) ? 16 : 0);
    int e = (kk & 7) | ((kk & 16) >> 1);
    return (ks * 32 + lane) * 16 + e;
}

// ---------------------------------------------------------------------------
// Fused GRU: 16 rows per block, 12 waves, each wave owns 2 of the 24 column
// tiles for BOTH the input-transform (K) and recurrent (R) matmuls.
// Weights register-resident for the whole t-loop; x_t and h are kept in LDS
// as f16 already packed in WMMA A-fragment order, so the matmul path does
// zero conversions: one contiguous v16h LDS load per fragment.
// If mask != null, fuses segment-sum pooling via atomicAdd into out[M,U].
// ---------------------------------------------------------------------------
__global__ void __launch_bounds__(GRU_THREADS, 1)
gru_fused(const float* __restrict__ seq,      // [N, 8, 128]
          const _Float16* __restrict__ wK,    // swizzled f16, FRAG_HALVES
          const _Float16* __restrict__ wR,    // swizzled f16, FRAG_HALVES
          const float* __restrict__ bias,     // [2, 384]
          int nRows,
          const unsigned char* __restrict__ mask,  // [N, nMol] bool or null
          int nMol,
          float* __restrict__ out) {          // mask? [nMol,128] : [N,128]
    __shared__ __align__(32) _Float16 axbuf[4 * 32 * 16];  // x_t, A-frag order (4KB)
    __shared__ __align__(32) _Float16 ahbuf[4 * 32 * 16];  // h,   A-frag order (4KB)
    __shared__ float lds_h [16][132];   // h fp32 (for z*h term)
    __shared__ float lds_xh[16][132];
    __shared__ float lds_rh[16][132];
    __shared__ float lds_g [16][260];   // cols 0..127 z-sum, 128..255 r-sum
    __shared__ int   s_mol [16];

    const int tid  = threadIdx.x;
    const int lane = tid & 31;
    const int w    = tid >> 5;          // wave 0..11
    const int rowBase = blockIdx.x * 16;

    // ---- register-resident swizzled weight fragments: 2 tiles x 4 ksteps x 2
    v16h Kf[2][4], Rf[2][4];
    float bx[2], br[2];
    #pragma unroll
    for (int j = 0; j < 2; ++j) {
        int ct = w * 2 + j;
        #pragma unroll
        for (int ks = 0; ks < 4; ++ks) {
            int f = ct * 4 + ks;
            Kf[j][ks] = *(const v16h*)(wK + (size_t)(f * 32 + lane) * 16);
            Rf[j][ks] = *(const v16h*)(wR + (size_t)(f * 32 + lane) * 16);
        }
        int n = ct * 16 + (lane & 15);
        bx[j] = bias[n];
        br[j] = bias[U3 + n];
    }

    // ---- h = 0 (both fp32 copy and f16 fragment buffer)
    for (int lin = tid; lin < 16 * 128; lin += GRU_THREADS)
        lds_h[lin >> 7][lin & 127] = 0.f;
    for (int lin = tid; lin < 1024; lin += GRU_THREADS)
        ((unsigned int*)ahbuf)[lin] = 0u;   // 2048 halves

    const int m16  = lane & 15;
    const int mofs = (lane >> 4) ? 8 : 0;

    for (int t = 0; t < T_HIST; ++t) {
        // ---- stage x_t (16x128 fp32) -> f16 A-fragment layout in LDS.
        // thread -> (ks, lane, hi): 2x float4 global load, one b128 LDS store.
        if (tid < 256) {
            int hi = tid & 1;
            int sl = (tid >> 1) & 31;      // fragment lane
            int ks = tid >> 6;             // 0..3
            int m  = sl & 15;
            int kstart = ks * 32 + ((sl >> 4) ? 8 : 0) + hi * 16;
            const float* p = seq + (((size_t)(rowBase + m) * T_HIST) + t) * D_IN + kstart;
            float4 f0 = *(const float4*)p;
            float4 f1 = *(const float4*)(p + 4);
            v8h hv;
            hv[0] = (_Float16)f0.x; hv[1] = (_Float16)f0.y;
            hv[2] = (_Float16)f0.z; hv[3] = (_Float16)f0.w;
            hv[4] = (_Float16)f1.x; hv[5] = (_Float16)f1.y;
            hv[6] = (_Float16)f1.z; hv[7] = (_Float16)f1.w;
            *(v8h*)&axbuf[(ks * 32 + sl) * 16 + hi * 8] = hv;
            if (t + 1 < T_HIST) __builtin_prefetch(p + D_IN, 0, 0);
        }
        __syncthreads();

        // ---- load A fragments: contiguous, pre-packed, no conversions
        v16h ax[4], ah[4];
        #pragma unroll
        for (int ks = 0; ks < 4; ++ks) {
            ax[ks] = *(const v16h*)&axbuf[(ks * 32 + lane) * 16];
            ah[ks] = *(const v16h*)&ahbuf[(ks * 32 + lane) * 16];
        }

        // ---- matmuls: x@K and h@R for this wave's 2 column tiles
        #pragma unroll
        for (int j = 0; j < 2; ++j) {
            v8f accx = {}; v8f acch = {};
            #pragma unroll
            for (int ks = 0; ks < 4; ++ks) {
                accx = __builtin_amdgcn_wmma_f32_16x16x32_f16(
                    false, ax[ks], false, Kf[j][ks], (short)0, accx, false, false);
                acch = __builtin_amdgcn_wmma_f32_16x16x32_f16(
                    false, ah[ks], false, Rf[j][ks], (short)0, acch, false, false);
            }
            int ct = w * 2 + j;
            int n  = ct * 16 + m16;
            if (ct < 16) {
                // z (cols 0..127) and r (cols 128..255): only the sum is needed
                #pragma unroll
                for (int v = 0; v < 8; ++v)
                    lds_g[v + mofs][n] = accx[v] + acch[v] + bx[j] + br[j];
            } else {
                int c = n - 256;     // h-candidate columns: keep xh, rh separate
                #pragma unroll
                for (int v = 0; v < 8; ++v) {
                    lds_xh[v + mofs][c] = accx[v] + bx[j];
                    lds_rh[v + mofs][c] = acch[v] + br[j];
                }
            }
        }
        __syncthreads();

        // ---- elementwise GRU update (reset_after); process column pairs so
        // the f16 fragment-layout h store is a single packed b32 per pair.
        for (int p = tid; p < 16 * 64; p += GRU_THREADS) {
            int m  = p >> 6;
            int c0 = (p & 63) * 2;
            float z0  = sigmoidf_(lds_g[m][c0]);
            float z1  = sigmoidf_(lds_g[m][c0 + 1]);
            float r0  = sigmoidf_(lds_g[m][128 + c0]);
            float r1  = sigmoidf_(lds_g[m][128 + c0 + 1]);
            float hh0 = tanhf(lds_xh[m][c0]     + r0 * lds_rh[m][c0]);
            float hh1 = tanhf(lds_xh[m][c0 + 1] + r1 * lds_rh[m][c0 + 1]);
            float hn0 = z0 * lds_h[m][c0]     + (1.f - z0) * hh0;
            float hn1 = z1 * lds_h[m][c0 + 1] + (1.f - z1) * hh1;
            lds_h[m][c0]     = hn0;
            lds_h[m][c0 + 1] = hn1;
            v2h pk; pk[0] = (_Float16)hn0; pk[1] = (_Float16)hn1;
            *(v2h*)&ahbuf[afrag_idx(m, c0)] = pk;   // even c -> 4B aligned pair
        }
        __syncthreads();
    }

    // ---- epilogue: fused segment-sum pooling or direct store
    if (mask != nullptr) {
        if (tid < 16) s_mol[tid] = 0;
        __syncthreads();
        if (tid < 256) {
            int m = tid >> 4;
            const unsigned char* mrow = mask + (size_t)(rowBase + m) * nMol;
            int partial = 0;
            for (int c = (tid & 15); c < nMol; c += 16)
                partial += (int)mrow[c] * c;     // exactly one true per row
            if (partial) atomicAdd(&s_mol[m], partial);
        }
        __syncthreads();
        for (int lin = tid; lin < 16 * 128; lin += GRU_THREADS) {
            int m = lin >> 7, c = lin & 127;
            atomicAdd(&out[(size_t)s_mol[m] * D_IN + c], lds_h[m][c]);
        }
    } else {
        for (int lin = tid; lin < 16 * 128; lin += GRU_THREADS) {
            int m = lin >> 7, c = lin & 127;
            out[(size_t)(rowBase + m) * D_IN + c] = lds_h[m][c];
        }
    }
}

// ---------------------------------------------------------------------------
// Tiny MLP head: one block (128 threads) per molecule.
// ---------------------------------------------------------------------------
__global__ void mlp_head(const float* __restrict__ yeb,
                         const float* __restrict__ yvb,
                         const float* __restrict__ yu,
                         const float* __restrict__ W1, const float* __restrict__ b1,
                         const float* __restrict__ W2, const float* __restrict__ b2,
                         const float* __restrict__ W3, const float* __restrict__ b3,
                         float* __restrict__ out) {
    __shared__ float in[U3];
    __shared__ float h1[128];
    __shared__ float red[128];
    int m = blockIdx.x, t = threadIdx.x;
    in[t]       = yeb[(size_t)m * 128 + t];
    in[128 + t] = yvb[(size_t)m * 128 + t];
    in[256 + t] = yu [(size_t)m * 128 + t];
    __syncthreads();
    float acc = b1[t];
    for (int k = 0; k < U3; ++k) acc += in[k] * W1[(size_t)k * 128 + t];
    float h1v = tanhf(acc);
    h1[t] = h1v;
    __syncthreads();
    acc = b2[t];
    for (int k = 0; k < 128; ++k) acc += h1[k] * W2[(size_t)k * 128 + t];
    float h2 = tanhf(acc);
    red[t] = h2 * W3[t];
    __syncthreads();
    for (int s = 64; s > 0; s >>= 1) {
        if (t < s) red[t] += red[t + s];
        __syncthreads();
    }
    if (t == 0) out[m] = red[0] + b3[0];
}

// ---------------------------------------------------------------------------
extern "C" void kernel_launch(void* const* d_in, const int* in_sizes, int n_in,
                              void* d_out, int out_size, void* d_ws, size_t ws_size,
                              hipStream_t stream) {
    const float* h_e = (const float*)d_in[0];
    const float* h_v = (const float*)d_in[1];
    const float* h_u = (const float*)d_in[2];
    const unsigned char* atom_in_mol = (const unsigned char*)d_in[3];
    const unsigned char* bond_in_mol = (const unsigned char*)d_in[4];
    const float* e_K = (const float*)d_in[5];
    const float* e_R = (const float*)d_in[6];
    const float* e_b = (const float*)d_in[7];
    const float* v_K = (const float*)d_in[8];
    const float* v_R = (const float*)d_in[9];
    const float* v_b = (const float*)d_in[10];
    const float* u_K = (const float*)d_in[11];
    const float* u_R = (const float*)d_in[12];
    const float* u_b = (const float*)d_in[13];
    const float* W1  = (const float*)d_in[14];
    const float* b1  = (const float*)d_in[15];
    const float* W2  = (const float*)d_in[16];
    const float* b2  = (const float*)d_in[17];
    const float* W3  = (const float*)d_in[18];
    const float* b3  = (const float*)d_in[19];

    const int N_BOND = 16384, N_ATOM = 16384, N_MOL = 512;

    // workspace layout
    _Float16* wsw = (_Float16*)d_ws;                               // 6 * 49152 halves
    size_t off = (size_t)6 * FRAG_HALVES * sizeof(_Float16);       // 589824 B
    float* y_u    = (float*)((char*)d_ws + off); off += (size_t)N_MOL * 128 * 4;
    float* ye_bar = (float*)((char*)d_ws + off); off += (size_t)N_MOL * 128 * 4;
    float* yv_bar = (float*)((char*)d_ws + off);

    // 1) weights -> f16, WMMA-fragment-swizzled
    swizzle_weights<<<(6 * FRAG_HALVES + 255) / 256, 256, 0, stream>>>(
        e_K, e_R, v_K, v_R, u_K, u_R, wsw);

    // 2) zero the pooling accumulators (ye_bar, yv_bar are contiguous)
    zero_f32<<<(2 * N_MOL * 128 + 255) / 256, 256, 0, stream>>>(ye_bar, 2 * N_MOL * 128);

    // 3) GRUs (u: direct store; e/v: fused atomic segment-sum pooling)
    gru_fused<<<N_MOL / 16, GRU_THREADS, 0, stream>>>(
        h_u, wsw + 4 * FRAG_HALVES, wsw + 5 * FRAG_HALVES, u_b,
        N_MOL, nullptr, N_MOL, y_u);
    gru_fused<<<N_BOND / 16, GRU_THREADS, 0, stream>>>(
        h_e, wsw + 0 * FRAG_HALVES, wsw + 1 * FRAG_HALVES, e_b,
        N_BOND, bond_in_mol, N_MOL, ye_bar);
    gru_fused<<<N_ATOM / 16, GRU_THREADS, 0, stream>>>(
        h_v, wsw + 2 * FRAG_HALVES, wsw + 3 * FRAG_HALVES, v_b,
        N_ATOM, atom_in_mol, N_MOL, yv_bar);

    // 4) MLP head -> d_out [512]
    mlp_head<<<N_MOL, 128, 0, stream>>>(
        ye_bar, yv_bar, y_u, W1, b1, W2, b2, W3, b3, (float*)d_out);
}
